// ChannelAttention_22926535426131
// MI455X (gfx1250) — matrix operand order repaired and checked by reading, not compiled, not run on hardware
//
#include <hip/hip_runtime.h>
#include <hip/hip_bf16.h>
#include <math.h>

// ---------------------------------------------------------------------------
// Types for CDNA5 WMMA (wave32, 16x16x32 bf16, fp32 accumulate)
// ---------------------------------------------------------------------------
typedef __attribute__((ext_vector_type(16))) __bf16 v16bf;
typedef __attribute__((ext_vector_type(8)))  __bf16 v8bf;
typedef __attribute__((ext_vector_type(4)))  __bf16 v4bf;
typedef __attribute__((ext_vector_type(8)))  float  v8f;
typedef __attribute__((ext_vector_type(4)))  float  v4f;
typedef __attribute__((ext_vector_type(4)))  int    v4i;

#define LDSS 40            // LDS row stride in bf16 elems (80B: 16B-aligned, bank-spread)
#define C_DIM 256
#define NVOX 32768         // W*D*H
#define ROWS_TOT 65536     // B * NVOX
#define PER_BRANCH 16777216 // ROWS_TOT * C_DIM
#define PER_BATCH 8388608   // NVOX * C_DIM

// ws layout (bytes)
#define OFF_WT   0          // bf16 Wt[4][256][256] (transposed weights)   524288
#define OFF_AF   524288     // f32 afold[4][256]                           4096
#define OFF_BF   528384     // f32 bfold[4][256]                           4096
#define OFF_M12  532480     // f32 m1/m2 [4][256][256]                     1048576
#define OFF_AFF  1581056    // bf16 affinity [2][256][256]                 262144
#define OFF_Y    1843200    // bf16 Q,K,J,V [4][65536][256]                134217728

// ---------------------------------------------------------------------------
// Async global->LDS copy (GLOBAL_LOAD_ASYNC_TO_LDS_B128, ASYNCcnt-tracked).
// Builtin takes typed int4 pointers (global src, LDS dst) per clang diagnostic.
// Falls back to a plain b128 copy if the builtin is unavailable.
// ---------------------------------------------------------------------------
#if __has_builtin(__builtin_amdgcn_global_load_async_to_lds_b128)
#define HAVE_ASYNC_LDS 1
#endif

typedef __attribute__((address_space(1))) v4i g_v4i;
typedef __attribute__((address_space(3))) v4i l_v4i;

__device__ __forceinline__ void async_copy16(const __bf16* gsrc, __bf16* ldst) {
#ifdef HAVE_ASYNC_LDS
  __builtin_amdgcn_global_load_async_to_lds_b128(
      (g_v4i*)(void*)gsrc, (l_v4i*)(void*)ldst, 0, 0);
#else
  *(v8bf*)ldst = *(const v8bf*)gsrc;
#endif
}

__device__ __forceinline__ void async_join() {
#ifdef HAVE_ASYNC_LDS
  asm volatile("s_wait_asynccnt 0x0" ::: "memory");
#endif
}

// ---------------------------------------------------------------------------
// Fragment gathers per CDNA5 ISA 16-bit A / B layouts (CDNA5 ISA §7.12.2)
// A (16x32, M x K): lane m=lane%16, h=lane/16; K = h*8+0..7 and 16+h*8+0..7
// B (32x16, K x N): lane n=lane%16, h=lane/16; K = h*8+0..7 and 16+h*8+0..7 (column n)
// Both gathers are contiguous along K when staged K-major with stride LDSS.
// ---------------------------------------------------------------------------
__device__ __forceinline__ v16bf frag_a(const __bf16* base) {
  const int lane = threadIdx.x & 31;
  const int m = lane & 15, h = lane >> 4;
  const __bf16* p = base + m * LDSS;
  v8bf lo = *(const v8bf*)(p + h * 8);
  v8bf hi = *(const v8bf*)(p + 16 + h * 8);
  v16bf r;
#pragma unroll
  for (int i = 0; i < 8; ++i) { r[i] = lo[i]; r[i + 8] = hi[i]; }
  return r;
}

__device__ __forceinline__ v16bf frag_b(const __bf16* base) {
  const int lane = threadIdx.x & 31;
  const int n = lane & 15, h = lane >> 4;
  const __bf16* p = base + n * LDSS;
  v8bf lo = *(const v8bf*)(p + h * 8);
  v8bf hi = *(const v8bf*)(p + 16 + h * 8);
  v16bf r;
#pragma unroll
  for (int i = 0; i < 8; ++i) { r[i] = lo[i]; r[i + 8] = hi[i]; }
  return r;
}

__device__ __forceinline__ v8f wmma_bf16(v16bf a, v16bf b, v8f c) {
  return __builtin_amdgcn_wmma_f32_16x16x32_bf16(
      /*neg_a=*/false, a, /*neg_b=*/false, b,
      /*c_mod=*/(short)0, c, /*reuse_a=*/false, /*reuse_b=*/false);
}

__device__ __forceinline__ v4bf cvt4(v4f f) {
  v4bf o;
#pragma unroll
  for (int i = 0; i < 4; ++i) o[i] = (__bf16)f[i];
  return o;
}

// ---------------------------------------------------------------------------
// Kernel 0: fold BN into per-channel scale/bias; transpose+convert W to bf16
//   afold = s * rsqrt(v + eps); bfold = afold*conv_b + o - afold*m
//   Wt[br][f][c] = (bf16) W[br][c][f]   (so B-tile staging is a row copy)
// ---------------------------------------------------------------------------
__global__ __launch_bounds__(256) void k_prep(
    const float* __restrict__ W, const float* __restrict__ cb,
    const float* __restrict__ s, const float* __restrict__ o,
    const float* __restrict__ m, const float* __restrict__ v,
    float* __restrict__ afold, float* __restrict__ bfold,
    __bf16* __restrict__ Wt) {
  int idx = blockIdx.x * 256 + threadIdx.x;       // grid 1024 -> 262144
  if (idx < 1024) {
    float a = s[idx] * rsqrtf(v[idx] + 1e-3f);
    afold[idx] = a;
    bfold[idx] = a * cb[idx] + o[idx] - a * m[idx];
  }
  int br = idx >> 16, c = (idx >> 8) & 255, f = idx & 255;
  Wt[(br << 16) + f * 256 + c] = (__bf16)W[idx];
}

__global__ __launch_bounds__(256) void k_zero(float* __restrict__ p) {
  p[blockIdx.x * 256 + threadIdx.x] = 0.f;        // grid 1024 -> 262144 floats
}

// ---------------------------------------------------------------------------
// Kernel 1: fused 4-branch conv(1x1x1)+BN+ReLU.
//   Y[br] (bf16, [65536][256]) = relu(afold * (X @ W[br]) + bfold)
// Tile: 64 rows x 64 cols, K=256 in chunks of 32; x tile shared by 4 branches.
// X staged via float4 loads + packed bf16 stores; Wt tiles via async b128 DMA.
// ---------------------------------------------------------------------------
__global__ __launch_bounds__(256) void k_branch(
    const float* __restrict__ X, const __bf16* __restrict__ Wt,
    const float* __restrict__ afold, const float* __restrict__ bfold,
    __bf16* __restrict__ Y) {
  __shared__ __align__(16) __bf16 As[64 * LDSS];
  __shared__ __align__(16) __bf16 Bs[4][64 * LDSS];
  const int tid = threadIdx.x;
  const int row0 = blockIdx.x * 64;
  const int n0 = blockIdx.y * 64;
  const int w = tid >> 5, wm = w & 3, wn = w >> 2;
  const int lane = tid & 31, nl = lane & 15, h = lane >> 4;

  v8f acc[4][2];
#pragma unroll
  for (int br = 0; br < 4; ++br)
#pragma unroll
    for (int t = 0; t < 2; ++t)
#pragma unroll
      for (int e = 0; e < 8; ++e) acc[br][t][e] = 0.f;

  for (int kc = 0; kc < 256; kc += 32) {
    // B: 4 branches x (64 n x 32 k) rows of Wt -- pure b128 copies -> async DMA
#pragma unroll
    for (int t = 0; t < 4; ++t) {
      int e8 = tid + t * 256;                     // 1024 vectors of 8 bf16
      int br = e8 >> 8, rem = e8 & 255;
      int n = rem >> 2, k8 = rem & 3;
      async_copy16(&Wt[(br << 16) + (n0 + n) * 256 + kc + k8 * 8],
                   &Bs[br][n * LDSS + k8 * 8]);
    }
    // A: 64x32 f32 -> bf16, float4 vectorized
#pragma unroll
    for (int t = 0; t < 2; ++t) {
      int e4 = tid + t * 256;                     // 512 float4 vectors
      int r = e4 >> 3, k4 = e4 & 7;
      v4f f = *(const v4f*)&X[(size_t)(row0 + r) * 256 + kc + k4 * 4];
      *(v4bf*)&As[r * LDSS + k4 * 4] = cvt4(f);
    }
    async_join();
    __syncthreads();
    v16bf a = frag_a(&As[(wm * 16) * LDSS]);
#pragma unroll
    for (int br = 0; br < 4; ++br)
#pragma unroll
      for (int t = 0; t < 2; ++t) {
        v16bf b = frag_b(&Bs[br][(wn * 32 + t * 16) * LDSS]);
        acc[br][t] = wmma_bf16(a, b, acc[br][t]);
      }
    __syncthreads();
  }
#pragma unroll
  for (int br = 0; br < 4; ++br)
#pragma unroll
    for (int t = 0; t < 2; ++t)
#pragma unroll
      for (int e = 0; e < 8; ++e) {
        int r = row0 + wm * 16 + h * 8 + e;       // C/D layout: M=e+h*8, N=lane%16
        int f = n0 + wn * 32 + t * 16 + nl;
        float val = afold[br * 256 + f] * acc[br][t][e] + bfold[br * 256 + f];
        val = val > 0.f ? val : 0.f;
        Y[(size_t)br * PER_BRANCH + (size_t)r * 256 + f] = (__bf16)val;
      }
}

// ---------------------------------------------------------------------------
// Kernel 2: split-K Gram GEMMs  m1 = K·Qᵀ, m2 = K·Jᵀ  (per batch, fp32 accum)
// Raw-reshape rows are contiguous 32768-elem slices -> both operands K-major.
// Double-buffered async DMA pipeline: prefetch chunk t+1 while WMMAs eat t.
// grid: x=split(32, K-slice 1024), y=2x2 128-blocks, z=batch*2+mat.
// Epilogue: global_atomic_add_f32 into m12 (zeroed by k_zero).
// ---------------------------------------------------------------------------
__global__ __launch_bounds__(256) void k_gram(
    const __bf16* __restrict__ Y, float* __restrict__ m12) {
  __shared__ __align__(16) __bf16 As[2][128 * LDSS];
  __shared__ __align__(16) __bf16 Bs[2][128 * LDSS];
  const int tid = threadIdx.x;
  const int z = blockIdx.z, b = z >> 1, mat = z & 1;
  const int i0 = (blockIdx.y & 1) * 128;
  const int j0 = (blockIdx.y >> 1) * 128;
  const size_t abase = (size_t)1 * PER_BRANCH + (size_t)b * PER_BATCH; // K branch
  const size_t bbase = (size_t)(mat ? 2 : 0) * PER_BRANCH + (size_t)b * PER_BATCH; // Q / J
  const int w = tid >> 5, wm = w & 3, wn = w >> 2;
  const int lane = tid & 31, nl = lane & 15, h = lane >> 4;

  v8f acc[2][4];
#pragma unroll
  for (int ti = 0; ti < 2; ++ti)
#pragma unroll
    for (int tj = 0; tj < 4; ++tj)
#pragma unroll
      for (int e = 0; e < 8; ++e) acc[ti][tj][e] = 0.f;

  const int kbase = blockIdx.x * 1024;
  auto stage = [&](int buf, int kc) {
#pragma unroll
    for (int t = 0; t < 2; ++t) {                 // 128x32 bf16 each, b128 DMA
      int e8 = tid + t * 256;
      int r = e8 >> 2, k8 = e8 & 3;
      async_copy16(&Y[abase + (size_t)(i0 + r) * NVOX + kc + k8 * 8],
                   &As[buf][r * LDSS + k8 * 8]);
      async_copy16(&Y[bbase + (size_t)(j0 + r) * NVOX + kc + k8 * 8],
                   &Bs[buf][r * LDSS + k8 * 8]);
    }
  };

  stage(0, kbase);
  async_join();
  __syncthreads();
  for (int ch = 0; ch < 32; ++ch) {
    const int cur = ch & 1;
    if (ch + 1 < 32) stage(cur ^ 1, kbase + (ch + 1) * 32);
    v16bf a0 = frag_a(&As[cur][(wm * 32) * LDSS]);
    v16bf a1 = frag_a(&As[cur][(wm * 32 + 16) * LDSS]);
#pragma unroll
    for (int tj = 0; tj < 4; ++tj) {
      v16bf bb = frag_b(&Bs[cur][(wn * 64 + tj * 16) * LDSS]);
      acc[0][tj] = wmma_bf16(a0, bb, acc[0][tj]);
      acc[1][tj] = wmma_bf16(a1, bb, acc[1][tj]);
    }
    async_join();                                 // prefetch landed
    __syncthreads();                              // all waves done read+write
  }
  float* M = m12 + (size_t)z * 65536;
#pragma unroll
  for (int ti = 0; ti < 2; ++ti)
#pragma unroll
    for (int tj = 0; tj < 4; ++tj)
#pragma unroll
      for (int e = 0; e < 8; ++e) {
        int i = i0 + wm * 32 + ti * 16 + h * 8 + e;
        int j = j0 + wn * 64 + tj * 16 + nl;
        atomicAdd(&M[i * 256 + j], acc[ti][tj][e]);
      }
}

// ---------------------------------------------------------------------------
// Kernel 3: affinity = sigmoid(m1 @ m2), output bf16 [2][256][256]
// grid: x=4 row-blocks of 64, y=batch. m2 staged transposed into LDS.
// ---------------------------------------------------------------------------
__global__ __launch_bounds__(256) void k_affinity(
    const float* __restrict__ m12, __bf16* __restrict__ aff) {
  __shared__ __align__(16) __bf16 As[64 * LDSS];
  __shared__ __align__(16) __bf16 Bs[256 * LDSS];
  const int tid = threadIdx.x;
  const int i0 = blockIdx.x * 64;
  const int b = blockIdx.y;
  const float* m1p = m12 + (size_t)(b * 2 + 0) * 65536;
  const float* m2p = m12 + (size_t)(b * 2 + 1) * 65536;
  const int w = tid >> 5, wm = w & 1, wn = w >> 1;
  const int lane = tid & 31, nl = lane & 15, h = lane >> 4;

  v8f acc[2][4];
#pragma unroll
  for (int ti = 0; ti < 2; ++ti)
#pragma unroll
    for (int tj = 0; tj < 4; ++tj)
#pragma unroll
      for (int e = 0; e < 8; ++e) acc[ti][tj][e] = 0.f;

  for (int kc = 0; kc < 256; kc += 32) {
#pragma unroll
    for (int t = 0; t < 2; ++t) {                 // A: 64x32 of m1, float4 loads
      int e4 = tid + t * 256;
      int r = e4 >> 3, k4 = e4 & 7;
      v4f f = *(const v4f*)&m1p[(i0 + r) * 256 + kc + k4 * 4];
      *(v4bf*)&As[r * LDSS + k4 * 4] = cvt4(f);
    }
#pragma unroll
    for (int t = 0; t < 32; ++t) {                // B: m2[k][n] -> Bs[n][k]
      int e = tid + t * 256;
      int k = e >> 8, n = e & 255;
      Bs[n * LDSS + k] = (__bf16)m2p[(kc + k) * 256 + n];
    }
    __syncthreads();
    v16bf a0 = frag_a(&As[(wm * 32) * LDSS]);
    v16bf a1 = frag_a(&As[(wm * 32 + 16) * LDSS]);
#pragma unroll
    for (int tj = 0; tj < 4; ++tj) {
      v16bf bb = frag_b(&Bs[(wn * 64 + tj * 16) * LDSS]);
      acc[0][tj] = wmma_bf16(a0, bb, acc[0][tj]);
      acc[1][tj] = wmma_bf16(a1, bb, acc[1][tj]);
    }
    __syncthreads();
  }
#pragma unroll
  for (int ti = 0; ti < 2; ++ti)
#pragma unroll
    for (int tj = 0; tj < 4; ++tj)
#pragma unroll
      for (int e = 0; e < 8; ++e) {
        int i = i0 + wm * 32 + ti * 16 + h * 8 + e;
        int j = wn * 64 + tj * 16 + nl;
        float val = acc[ti][tj][e];
        val = 1.f / (1.f + __expf(-val));
        aff[(size_t)b * 65536 + i * 256 + j] = (__bf16)val;
      }
}

// ---------------------------------------------------------------------------
// Kernel 4: out = gamma * (affinity @ V) + x    (fused residual epilogue)
// grid: x=256 col-blocks of 128 (nn), y=2 row-blocks of 128 (i), z=batch.
// Affinity rows async b128 DMA; V staged transposed via v8bf reads + b16 LDS.
// ---------------------------------------------------------------------------
__global__ __launch_bounds__(256) void k_out(
    const __bf16* __restrict__ aff, const __bf16* __restrict__ Y,
    const float* __restrict__ X, const float* __restrict__ gammap,
    float* __restrict__ out) {
  __shared__ __align__(16) __bf16 As[128 * LDSS];
  __shared__ __align__(16) __bf16 Bs[128 * LDSS];
  const int tid = threadIdx.x;
  const int nn0 = blockIdx.x * 128;
  const int i0 = blockIdx.y * 128;
  const int b = blockIdx.z;
  const __bf16* affp = aff + (size_t)b * 65536;
  const __bf16* Vp = Y + (size_t)3 * PER_BRANCH + (size_t)b * PER_BATCH;
  const int w = tid >> 5, wm = w & 3, wn = w >> 2;
  const int lane = tid & 31, nl = lane & 15, h = lane >> 4;

  v8f acc[2][4];
#pragma unroll
  for (int ti = 0; ti < 2; ++ti)
#pragma unroll
    for (int tj = 0; tj < 4; ++tj)
#pragma unroll
      for (int e = 0; e < 8; ++e) acc[ti][tj][e] = 0.f;

  for (int kc = 0; kc < 256; kc += 32) {
#pragma unroll
    for (int t = 0; t < 2; ++t) {                 // A: 128x32 bf16 rows of affinity
      int e8 = tid + t * 256;
      int r = e8 >> 2, k8 = e8 & 3;
      async_copy16(&affp[(size_t)(i0 + r) * 256 + kc + k8 * 8],
                   &As[r * LDSS + k8 * 8]);
    }
#pragma unroll
    for (int t = 0; t < 2; ++t) {                 // B: V[j][nn] -> Bs[nn][j], b128 reads
      int e8 = tid + t * 256;                     // 512 vectors over 32k x 128nn
      int k = e8 >> 4, n8 = e8 & 15;
      v8bf vv = *(const v8bf*)&Vp[(size_t)(kc + k) * NVOX + nn0 + n8 * 8];
#pragma unroll
      for (int i = 0; i < 8; ++i) Bs[(n8 * 8 + i) * LDSS + k] = vv[i];
    }
    async_join();
    __syncthreads();
    v16bf a0 = frag_a(&As[(wm * 32) * LDSS]);
    v16bf a1 = frag_a(&As[(wm * 32 + 16) * LDSS]);
#pragma unroll
    for (int tj = 0; tj < 4; ++tj) {
      v16bf bb = frag_b(&Bs[(wn * 64 + tj * 16) * LDSS]);
      acc[0][tj] = wmma_bf16(a0, bb, acc[0][tj]);
      acc[1][tj] = wmma_bf16(a1, bb, acc[1][tj]);
    }
    __syncthreads();
  }
  const float g = gammap[0];
#pragma unroll
  for (int ti = 0; ti < 2; ++ti)
#pragma unroll
    for (int tj = 0; tj < 4; ++tj)
#pragma unroll
      for (int e = 0; e < 8; ++e) {
        int i = i0 + wm * 32 + ti * 16 + h * 8 + e;
        int nn = nn0 + wn * 64 + tj * 16 + nl;
        size_t idx = (size_t)b * PER_BATCH + (size_t)i * NVOX + nn;
        out[idx] = g * acc[ti][tj][e] + X[idx];
      }
}

// ---------------------------------------------------------------------------
extern "C" void kernel_launch(void* const* d_in, const int* in_sizes, int n_in,
                              void* d_out, int out_size, void* d_ws, size_t ws_size,
                              hipStream_t stream) {
  (void)in_sizes; (void)n_in; (void)out_size; (void)ws_size;
  const float* X  = (const float*)d_in[0];
  const float* W  = (const float*)d_in[1];
  const float* cb = (const float*)d_in[2];
  const float* s  = (const float*)d_in[3];
  const float* o  = (const float*)d_in[4];
  const float* m  = (const float*)d_in[5];
  const float* v  = (const float*)d_in[6];
  const float* g  = (const float*)d_in[7];
  float* out = (float*)d_out;

  char* ws = (char*)d_ws;
  __bf16* Wt    = (__bf16*)(ws + OFF_WT);
  float*  afold = (float*)(ws + OFF_AF);
  float*  bfold = (float*)(ws + OFF_BF);
  float*  m12   = (float*)(ws + OFF_M12);
  __bf16* aff   = (__bf16*)(ws + OFF_AFF);
  __bf16* Y     = (__bf16*)(ws + OFF_Y);

  k_prep<<<1024, 256, 0, stream>>>(W, cb, s, o, m, v, afold, bfold, Wt);
  k_zero<<<1024, 256, 0, stream>>>(m12);
  k_branch<<<dim3(1024, 4), 256, 0, stream>>>(X, Wt, afold, bfold, Y);
  k_gram<<<dim3(32, 4, 4), 256, 0, stream>>>(Y, m12);
  k_affinity<<<dim3(4, 2), 256, 0, stream>>>(m12, aff);
  k_out<<<dim3(256, 2, 2), 256, 0, stream>>>(aff, Y, X, g, out);
}
